// SCVDRNNAgent_14757507629759
// MI455X (gfx1250) — matrix operand order repaired and checked by reading, not compiled, not run on hardware
//
#include <hip/hip_runtime.h>

// Problem constants (from reference)
#define NEg   32   // NE entities
#define NAg   16   // NA agents
#define INDIM 96
#define HDIM  64   // H
#define ADIM  64   // AD
#define NHEAD 4
#define NACTk 20
#define TZk   32

typedef __attribute__((ext_vector_type(16))) _Float16 v16h;
typedef __attribute__((ext_vector_type(8)))  _Float16 v8h;
typedef __attribute__((ext_vector_type(8)))  float    v8f;

// ---- weight-fragment workspace layout (one 1KB block per B fragment) ----
// fragment = 32 lanes x 16 halves, stored lane-major (32B per lane, coalesced)
#define OFF_W1   0     // 12 frags (KT=3)
#define OFF_Wq   12    // 8  (KT=2)
#define OFF_Wk   20    // 8
#define OFF_Wv   28    // 8
#define OFF_Wih  36    // 48 (KT=4)
#define OFF_Whh  84    // 24 (KT=2)
#define OFF_Wqa  108   // 6  (KT=3)
#define OFF_Wact 114   // 4  (KT=1)
#define OFF_Wdq  118   // 8  (KT=2)
#define OFF_Wdk  126   // 8
#define OFF_Wdv  134   // 8
#define OFF_Wqo  142   // 10 (KT=5)
#define NFRAG    152
#define WS_NEEDED ((size_t)NFRAG * 512 * sizeof(_Float16))

__device__ __forceinline__ v8f wmma16(v16h a, v16h b, v8f c) {
  return __builtin_amdgcn_wmma_f32_16x16x32_f16(false, a, false, b, (short)0, c,
                                                false, false);
}

// ---- fragment builders (layouts per CDNA5 ISA 7.12.2, wave32) ----
// A 16x32 f16 from LDS: lane=(hi<<4)|m ; element i -> K = (i<8 ? i : i+8) + 8*hi
__device__ __forceinline__ v16h frag_a_lds(const _Float16* S, int ld, int m0, int k0) {
  int lane = threadIdx.x & 31;
  const _Float16* row = S + (m0 + (lane & 15)) * ld;
  int kb = k0 + ((lane >> 4) << 3);
  v8h lo = *(const v8h*)(row + kb);        // 16B-aligned -> ds_load_b128
  v8h hi = *(const v8h*)(row + kb + 16);
  v16h a;
#pragma unroll
  for (int i = 0; i < 8; ++i) { a[i] = lo[i]; a[i + 8] = hi[i]; }
  return a;
}

// A fragment broadcasting LDS row 0 to all M (for aux_emb)
__device__ __forceinline__ v16h frag_a_lds_row0(const _Float16* S, int k0) {
  int kb = k0 + (((threadIdx.x & 31) >> 4) << 3);
  v8h lo = *(const v8h*)(S + kb);
  v8h hi = *(const v8h*)(S + kb + 16);
  v16h a;
#pragma unroll
  for (int i = 0; i < 8; ++i) { a[i] = lo[i]; a[i + 8] = hi[i]; }
  return a;
}

// A fragment from global f32 row-major [M][ld] (K contiguous)
__device__ __forceinline__ v16h frag_a_gf32(const float* G, int ld, int m0, int k0) {
  int lane = threadIdx.x & 31;
  const float* row = G + (m0 + (lane & 15)) * ld;
  int kb = k0 + ((lane >> 4) << 3);
  v16h a;
#pragma unroll
  for (int i = 0; i < 8; ++i) a[i] = (_Float16)row[kb + i];
#pragma unroll
  for (int i = 0; i < 8; ++i) a[i + 8] = (_Float16)row[kb + 16 + i];
  return a;
}

// B 32x16 = W^T tile built on the fly from f32 weights (fallback / preamble)
__device__ __forceinline__ v16h frag_b_wT(const float* W, int ldk, int n0, int k0,
                                          int Nv, int Kv) {
  int lane = threadIdx.x & 31;
  int n = n0 + (lane & 15);
  int kb = k0 + ((lane >> 4) << 3);
  const float* row = W + n * ldk;
  v16h b;
#pragma unroll
  for (int i = 0; i < 16; ++i) {
    int k = kb + (i < 8 ? i : i + 8);
    float w = (n < Nv && k < Kv) ? row[k] : 0.f;
    b[i] = (_Float16)w;
  }
  return b;
}

// B fragment: pre-swizzled f16 from workspace (PREP) or on-the-fly f32
template <bool PREP>
__device__ __forceinline__ v16h frag_b(const _Float16* __restrict__ ws, int fidx,
                                       const float* W, int ldk, int n0, int k0,
                                       int Nv, int Kv) {
  if constexpr (PREP) {
    const _Float16* p = ws + (size_t)fidx * 512 + (threadIdx.x & 31) * 16;
    return *(const v16h*)p;  // 32B/lane, wave reads 1KB contiguous
  } else {
    return frag_b_wT(W, ldk, n0, k0, Nv, Kv);
  }
}

// C init with per-N bias (element r -> (M=mb+r, N=n); N constant per lane)
__device__ __forceinline__ v8f frag_c_bias(const float* bias, int n0, int Nv) {
  int n = n0 + (threadIdx.x & 15);
  float bv = (bias != nullptr && n < Nv) ? bias[n] : 0.f;
  v8f c = {bv, bv, bv, bv, bv, bv, bv, bv};
  return c;
}

__device__ __forceinline__ void store_c_f16(v8f c, _Float16* S, int ld, int m0, int n0,
                                            bool relu) {
  int lane = threadIdx.x & 31;
  int n = n0 + (lane & 15);
  int mb = m0 + ((lane >> 4) << 3);
#pragma unroll
  for (int r = 0; r < 8; ++r) {
    float x = c[r];
    if (relu) x = fmaxf(x, 0.f);
    S[(mb + r) * ld + n] = (_Float16)x;
  }
}

__device__ __forceinline__ void store_c_f32(v8f c, float* S, int ld, int m0, int n0) {
  int lane = threadIdx.x & 31;
  int n = n0 + (lane & 15);
  int mb = m0 + ((lane >> 4) << 3);
#pragma unroll
  for (int r = 0; r < 8; ++r) S[(mb + r) * ld + n] = c[r];
}

__device__ __forceinline__ float sigmoidf_(float x) { return 1.f / (1.f + __expf(-x)); }

// ---- preamble: convert all weight tiles to f16 B-fragment layout in d_ws ----
__global__ __launch_bounds__(32) void prep_wfrags(
    const float* W1, const float* Wq, const float* Wk, const float* Wv,
    const float* Wih, const float* Whh, const float* Wqa, const float* Wact,
    const float* Wdq, const float* Wdk, const float* Wdv, const float* Wqo,
    _Float16* __restrict__ ws) {
  int f = blockIdx.x;
  const float* W;
  int ldk, Nv, Kv, KT, r = f;
  if (f < OFF_Wq)        { W = W1;   ldk = 96;  Nv = 64;  Kv = 96;  KT = 3; r -= OFF_W1; }
  else if (f < OFF_Wk)   { W = Wq;   ldk = 64;  Nv = 64;  Kv = 64;  KT = 2; r -= OFF_Wq; }
  else if (f < OFF_Wv)   { W = Wk;   ldk = 64;  Nv = 64;  Kv = 64;  KT = 2; r -= OFF_Wk; }
  else if (f < OFF_Wih)  { W = Wv;   ldk = 64;  Nv = 64;  Kv = 64;  KT = 2; r -= OFF_Wv; }
  else if (f < OFF_Whh)  { W = Wih;  ldk = 128; Nv = 192; Kv = 128; KT = 4; r -= OFF_Wih; }
  else if (f < OFF_Wqa)  { W = Whh;  ldk = 64;  Nv = 192; Kv = 64;  KT = 2; r -= OFF_Whh; }
  else if (f < OFF_Wact) { W = Wqa;  ldk = 96;  Nv = 20;  Kv = 96;  KT = 3; r -= OFF_Wqa; }
  else if (f < OFF_Wdq)  { W = Wact; ldk = 21;  Nv = 64;  Kv = 21;  KT = 1; r -= OFF_Wact; }
  else if (f < OFF_Wdk)  { W = Wdq;  ldk = 64;  Nv = 64;  Kv = 64;  KT = 2; r -= OFF_Wdq; }
  else if (f < OFF_Wdv)  { W = Wdk;  ldk = 64;  Nv = 64;  Kv = 64;  KT = 2; r -= OFF_Wdk; }
  else if (f < OFF_Wqo)  { W = Wdv;  ldk = 64;  Nv = 64;  Kv = 64;  KT = 2; r -= OFF_Wdv; }
  else                   { W = Wqo;  ldk = 160; Nv = 20;  Kv = 160; KT = 5; r -= OFF_Wqo; }
  int nt = r / KT, kt = r % KT;
  v16h b = frag_b_wT(W, ldk, nt * 16, kt * 32, Nv, Kv);
  _Float16* p = ws + (size_t)f * 512 + (threadIdx.x & 31) * 16;
#pragma unroll
  for (int i = 0; i < 16; ++i) p[i] = b[i];
}

// ---- main fused forward: one wave32 workgroup per batch element ----
template <bool PREP>
__global__ __launch_bounds__(32) void scvd_agent_fwd(
    const float* __restrict__ inputs, const float* __restrict__ onehot,
    const float* __restrict__ task_z, const float* __restrict__ hidden,
    const int* __restrict__ obs_mask, const int* __restrict__ scen_mask,
    const float* __restrict__ W1, const float* __restrict__ b1,
    const float* __restrict__ Wq, const float* __restrict__ bq,
    const float* __restrict__ Wk, const float* __restrict__ bk,
    const float* __restrict__ Wv, const float* __restrict__ bv,
    const float* __restrict__ Wih, const float* __restrict__ Whh,
    const float* __restrict__ bih, const float* __restrict__ bhh,
    const float* __restrict__ Wqa, const float* __restrict__ bqa,
    const float* __restrict__ Wact,
    const float* __restrict__ Wdq, const float* __restrict__ bdq,
    const float* __restrict__ Wdk, const float* __restrict__ bdk,
    const float* __restrict__ Wdv, const float* __restrict__ bdv,
    const float* __restrict__ Wqo, const float* __restrict__ bqo,
    const _Float16* __restrict__ wsf, float* __restrict__ out, int BS) {
  __shared__ alignas(16) _Float16 sE[32 * 64];
  __shared__ alignas(16) _Float16 sQ[16 * 64];
  __shared__ alignas(16) _Float16 sK[32 * 64];
  __shared__ alignas(16) _Float16 sV[32 * 64];
  __shared__ alignas(16) _Float16 sX[16 * 128];
  __shared__ float    sW[4 * 16 * 32];
  __shared__ float    sG[16 * 192];
  __shared__ float    sG2[16 * 64];
  __shared__ alignas(16) _Float16 sHh[16 * 64];
  __shared__ alignas(16) _Float16 sTZ[16 * 32];
  __shared__ float    sQA[16 * 20];
  __shared__ alignas(16) _Float16 sSh[16 * 32];
  __shared__ alignas(16) _Float16 sAE[16 * 64];
  __shared__ alignas(16) _Float16 sQd[16 * 64];
  __shared__ alignas(16) _Float16 sKd[16 * 64];
  __shared__ alignas(16) _Float16 sVd[16 * 64];
  __shared__ alignas(16) _Float16 sEx[16 * 64];
  __shared__ int      sScen[16];

  const int b = blockIdx.x;
  const int lane = threadIdx.x;
  const float* xin = inputs + (size_t)b * NEg * INDIM;
  const float* hprev = hidden + (size_t)b * NAg * HDIM;

  if (lane < 16) sScen[lane] = scen_mask[b * NAg + lane];
  for (int t = lane; t < NAg * TZk; t += 32)
    sTZ[t] = (_Float16)task_z[(size_t)b * NAg * TZk + t];
  __syncthreads();

  // ---- 1) embed = relu(inputs @ W1^T + b1) ----
  {
    v16h aIn[2][3];
#pragma unroll
    for (int mt = 0; mt < 2; ++mt)
#pragma unroll
      for (int kt = 0; kt < 3; ++kt) aIn[mt][kt] = frag_a_gf32(xin, 96, mt * 16, kt * 32);
    for (int nt = 0; nt < 4; ++nt) {
      v16h B0 = frag_b<PREP>(wsf, OFF_W1 + nt * 3 + 0, W1, 96, nt * 16, 0, 64, 96);
      v16h B1 = frag_b<PREP>(wsf, OFF_W1 + nt * 3 + 1, W1, 96, nt * 16, 32, 64, 96);
      v16h B2 = frag_b<PREP>(wsf, OFF_W1 + nt * 3 + 2, W1, 96, nt * 16, 64, 64, 96);
#pragma unroll
      for (int mt = 0; mt < 2; ++mt) {
        v8f c = frag_c_bias(b1, nt * 16, 64);
        c = wmma16(aIn[mt][0], B0, c);
        c = wmma16(aIn[mt][1], B1, c);
        c = wmma16(aIn[mt][2], B2, c);
        store_c_f16(c, sE, 64, mt * 16, nt * 16, true);
      }
    }
  }
  __syncthreads();

  // ---- 2) q (agents only), k, v over all entities ----
  {
    v16h aE[2][2];
#pragma unroll
    for (int mt = 0; mt < 2; ++mt)
#pragma unroll
      for (int kt = 0; kt < 2; ++kt) aE[mt][kt] = frag_a_lds(sE, 64, mt * 16, kt * 32);
    for (int nt = 0; nt < 4; ++nt) {
      v16h Bq0 = frag_b<PREP>(wsf, OFF_Wq + nt * 2 + 0, Wq, 64, nt * 16, 0, 64, 64);
      v16h Bq1 = frag_b<PREP>(wsf, OFF_Wq + nt * 2 + 1, Wq, 64, nt * 16, 32, 64, 64);
      v8f cq = frag_c_bias(bq, nt * 16, 64);
      cq = wmma16(aE[0][0], Bq0, cq);
      cq = wmma16(aE[0][1], Bq1, cq);
      store_c_f16(cq, sQ, 64, 0, nt * 16, false);
      v16h Bk0 = frag_b<PREP>(wsf, OFF_Wk + nt * 2 + 0, Wk, 64, nt * 16, 0, 64, 64);
      v16h Bk1 = frag_b<PREP>(wsf, OFF_Wk + nt * 2 + 1, Wk, 64, nt * 16, 32, 64, 64);
      v16h Bv0 = frag_b<PREP>(wsf, OFF_Wv + nt * 2 + 0, Wv, 64, nt * 16, 0, 64, 64);
      v16h Bv1 = frag_b<PREP>(wsf, OFF_Wv + nt * 2 + 1, Wv, 64, nt * 16, 32, 64, 64);
#pragma unroll
      for (int mt = 0; mt < 2; ++mt) {
        v8f ck = frag_c_bias(bk, nt * 16, 64);
        v8f cv = frag_c_bias(bv, nt * 16, 64);
        ck = wmma16(aE[mt][0], Bk0, ck);
        ck = wmma16(aE[mt][1], Bk1, ck);
        cv = wmma16(aE[mt][0], Bv0, cv);
        cv = wmma16(aE[mt][1], Bv1, cv);
        store_c_f16(ck, sK, 64, mt * 16, nt * 16, false);
        store_c_f16(cv, sV, 64, mt * 16, nt * 16, true);
      }
    }
  }
  __syncthreads();

  // ---- 3) masked attention over entities (K=HD=16 -> VALU) ----
  {
    const int* om = obs_mask + (size_t)b * NEg * NEg;
    for (int t = lane; t < NHEAD * NAg * NEg; t += 32) {
      int h = t >> 9, a = (t >> 5) & 15, e = t & 31;
      const _Float16* qr = sQ + a * 64 + h * 16;
      const _Float16* kr = sK + e * 64 + h * 16;
      float s = 0.f;
#pragma unroll
      for (int d = 0; d < 16; ++d) s += (float)qr[d] * (float)kr[d];
      s *= 0.25f;  // 1/sqrt(HD)
      bool msk = (a == e) || (om[a * 32 + e] != 0);
      sW[t] = msk ? -1e9f : s;
    }
    __syncthreads();
    for (int t = lane; t < NHEAD * NAg; t += 32) {
      float* row = sW + t * 32;
      float mx = -1e30f;
      for (int e = 0; e < 32; ++e) mx = fmaxf(mx, row[e]);
      if (mx <= -1e8f) {
        for (int e = 0; e < 32; ++e) row[e] = 0.f;
      } else {
        float ssum = 0.f;
        for (int e = 0; e < 32; ++e) { float w = __expf(row[e] - mx); row[e] = w; ssum += w; }
        float inv = 1.f / ssum;
        for (int e = 0; e < 32; ++e) row[e] *= inv;
      }
    }
    __syncthreads();
    for (int t = lane; t < NAg * ADIM; t += 32) {
      int a = t >> 6, j = t & 63, h = j >> 4;
      const float* wr = sW + (h * 16 + a) * 32;
      float acc = 0.f;
      for (int e = 0; e < 32; ++e) acc += wr[e] * (float)sV[e * 64 + j];
      sX[a * 128 + 64 + j] = (_Float16)acc;
      sX[a * 128 + j] = sE[a * 64 + j];
    }
    __syncthreads();
  }

  // ---- 4) GRU ----
  {
    v16h aX0 = frag_a_lds(sX, 128, 0, 0),  aX1 = frag_a_lds(sX, 128, 0, 32);
    v16h aX2 = frag_a_lds(sX, 128, 0, 64), aX3 = frag_a_lds(sX, 128, 0, 96);
    v16h aH0 = frag_a_gf32(hprev, 64, 0, 0), aH1 = frag_a_gf32(hprev, 64, 0, 32);
    for (int nt = 0; nt < 12; ++nt) {
      int n0 = nt * 16;
      v8f c = frag_c_bias(bih, n0, 192);
      c = wmma16(aX0, frag_b<PREP>(wsf, OFF_Wih + nt * 4 + 0, Wih, 128, n0, 0, 192, 128), c);
      c = wmma16(aX1, frag_b<PREP>(wsf, OFF_Wih + nt * 4 + 1, Wih, 128, n0, 32, 192, 128), c);
      c = wmma16(aX2, frag_b<PREP>(wsf, OFF_Wih + nt * 4 + 2, Wih, 128, n0, 64, 192, 128), c);
      c = wmma16(aX3, frag_b<PREP>(wsf, OFF_Wih + nt * 4 + 3, Wih, 128, n0, 96, 192, 128), c);
      v16h Bh0 = frag_b<PREP>(wsf, OFF_Whh + nt * 2 + 0, Whh, 64, n0, 0, 192, 64);
      v16h Bh1 = frag_b<PREP>(wsf, OFF_Whh + nt * 2 + 1, Whh, 64, n0, 32, 192, 64);
      if (nt < 8) {  // r,z gates: gi+gh combined
        float bh = bhh[n0 + (lane & 15)];
#pragma unroll
        for (int r = 0; r < 8; ++r) c[r] += bh;
        c = wmma16(aH0, Bh0, c);
        c = wmma16(aH1, Bh1, c);
        store_c_f32(c, sG, 192, 0, n0);
      } else {       // n gate: gi and gh kept separate (r multiplies gh only)
        store_c_f32(c, sG, 192, 0, n0);
        v8f c2 = frag_c_bias(bhh, n0, 192);
        c2 = wmma16(aH0, Bh0, c2);
        c2 = wmma16(aH1, Bh1, c2);
        store_c_f32(c2, sG2, 64, 0, n0 - 128);
      }
    }
  }
  __syncthreads();

  const size_t Osz = (size_t)BS * NAg * NACTk;
  const size_t obase = (size_t)b * NAg * NACTk;
  {  // gates + mask + write h (output slot 4)
    float* outh = out + 4 * Osz + (size_t)b * NAg * HDIM;
    for (int t = lane; t < NAg * HDIM; t += 32) {
      int m = t >> 6, j = t & 63;
      float rg = sigmoidf_(sG[m * 192 + j]);
      float zg = sigmoidf_(sG[m * 192 + 64 + j]);
      float ng = tanhf(sG[m * 192 + 128 + j] + rg * sG2[m * 64 + j]);
      float hv = (1.f - zg) * ng + zg * hprev[m * 64 + j];
      if (sScen[m] != 0) hv = 0.f;
      sHh[t] = (_Float16)hv;
      outh[t] = hv;
    }
  }
  __syncthreads();

  // ---- 5) q_alone = [h | task_z] @ Wqa^T + bqa (N padded 20->32) ----
  {
    v16h aH0 = frag_a_lds(sHh, 64, 0, 0), aH1 = frag_a_lds(sHh, 64, 0, 32);
    v16h aTZ = frag_a_lds(sTZ, 32, 0, 0);
    for (int nt = 0; nt < 2; ++nt) {
      int n0 = nt * 16;
      v8f c = frag_c_bias(bqa, n0, 20);
      c = wmma16(aH0, frag_b<PREP>(wsf, OFF_Wqa + nt * 3 + 0, Wqa, 96, n0, 0, 20, 96), c);
      c = wmma16(aH1, frag_b<PREP>(wsf, OFF_Wqa + nt * 3 + 1, Wqa, 96, n0, 32, 20, 96), c);
      c = wmma16(aTZ, frag_b<PREP>(wsf, OFF_Wqa + nt * 3 + 2, Wqa, 96, n0, 64, 20, 96), c);
      int n = n0 + (lane & 15);
      int mb = (lane >> 4) << 3;
      if (n < 20) {
#pragma unroll
        for (int r = 0; r < 8; ++r) {
          int m = mb + r;
          float v = c[r];
          if (sScen[m] != 0) v = 0.f;
          sQA[m * 20 + n] = v;
          out[2 * Osz + obase + m * 20 + n] = v;  // q_alone
        }
      }
    }
  }
  __syncthreads();

  // ---- 6) decoder ----
  {
    const float* oh = onehot + (size_t)b * NAg * NACTk;
    for (int t = lane; t < NAg * 32; t += 32) {
      int m = t >> 5, cc = t & 31;
      float v = 0.f;
      if (m == 0 && cc == 0) v = 1.f;
      else if (m >= 1 && cc >= 1 && cc <= NACTk) v = oh[(m - 1) * NACTk + (cc - 1)];
      sSh[t] = (_Float16)v;
    }
  }
  __syncthreads();
  {
    v16h aSh = frag_a_lds(sSh, 32, 0, 0);
    for (int nt = 0; nt < 4; ++nt) {  // act_emb = relu(shifted @ Wact^T), K pad 21->32
      v8f c = frag_c_bias(nullptr, 0, 0);
      c = wmma16(aSh, frag_b<PREP>(wsf, OFF_Wact + nt, Wact, 21, nt * 16, 0, 64, 21), c);
      store_c_f16(c, sAE, 64, 0, nt * 16, true);
    }
  }
  __syncthreads();
  {
    v16h ah0 = frag_a_lds(sHh, 64, 0, 0), ah1 = frag_a_lds(sHh, 64, 0, 32);
    v16h ae0 = frag_a_lds(sAE, 64, 0, 0), ae1 = frag_a_lds(sAE, 64, 0, 32);
    for (int nt = 0; nt < 4; ++nt) {
      v8f cq = frag_c_bias(bdq, nt * 16, 64);
      v8f ck = frag_c_bias(bdk, nt * 16, 64);
      v8f cv = frag_c_bias(bdv, nt * 16, 64);
      cq = wmma16(ah0, frag_b<PREP>(wsf, OFF_Wdq + nt * 2 + 0, Wdq, 64, nt * 16, 0, 64, 64), cq);
      cq = wmma16(ah1, frag_b<PREP>(wsf, OFF_Wdq + nt * 2 + 1, Wdq, 64, nt * 16, 32, 64, 64), cq);
      ck = wmma16(ae0, frag_b<PREP>(wsf, OFF_Wdk + nt * 2 + 0, Wdk, 64, nt * 16, 0, 64, 64), ck);
      ck = wmma16(ae1, frag_b<PREP>(wsf, OFF_Wdk + nt * 2 + 1, Wdk, 64, nt * 16, 32, 64, 64), ck);
      cv = wmma16(ae0, frag_b<PREP>(wsf, OFF_Wdv + nt * 2 + 0, Wdv, 64, nt * 16, 0, 64, 64), cv);
      cv = wmma16(ae1, frag_b<PREP>(wsf, OFF_Wdv + nt * 2 + 1, Wdv, 64, nt * 16, 32, 64, 64), cv);
      store_c_f16(cq, sQd, 64, 0, nt * 16, false);
      store_c_f16(ck, sKd, 64, 0, nt * 16, false);
      store_c_f16(cv, sVd, 64, 0, nt * 16, true);
    }
  }
  __syncthreads();
  for (int t = lane; t < NHEAD * NAg * NAg; t += 32) {  // causal+pair masked scores
    int h = t >> 8, a = (t >> 4) & 15, n = t & 15;
    const _Float16* qr = sQd + a * 64 + h * 16;
    const _Float16* kr = sKd + n * 64 + h * 16;
    float s = 0.f;
#pragma unroll
    for (int d = 0; d < 16; ++d) s += (float)qr[d] * (float)kr[d];
    s *= 0.25f;
    bool msk = (n > a) || (sScen[a] != 0) || (sScen[n] != 0);
    sW[t] = msk ? -1e9f : s;
  }
  __syncthreads();
  for (int t = lane; t < NHEAD * NAg; t += 32) {
    float* row = sW + t * 16;
    float mx = -1e30f;
    for (int e = 0; e < 16; ++e) mx = fmaxf(mx, row[e]);
    if (mx <= -1e8f) {
      for (int e = 0; e < 16; ++e) row[e] = 0.f;
    } else {
      float ssum = 0.f;
      for (int e = 0; e < 16; ++e) { float w = __expf(row[e] - mx); row[e] = w; ssum += w; }
      float inv = 1.f / ssum;
      for (int e = 0; e < 16; ++e) row[e] *= inv;
    }
  }
  __syncthreads();
  for (int t = lane; t < NAg * ADIM; t += 32) {
    int a = t >> 6, j = t & 63, h = j >> 4;
    const float* wr = sW + (h * 16 + a) * 16;
    float acc = 0.f;
    for (int e = 0; e < 16; ++e) acc += wr[e] * (float)sVd[e * 64 + j];
    sEx[t] = (_Float16)acc;
  }
  __syncthreads();

  // ---- 7) all_qi and aux_q share Wqo B-fragments ----
  {
    v16h ah0 = frag_a_lds(sHh, 64, 0, 0), ah1 = frag_a_lds(sHh, 64, 0, 32);
    v16h ax0 = frag_a_lds(sEx, 64, 0, 0), ax1 = frag_a_lds(sEx, 64, 0, 32);
    v16h ar0 = frag_a_lds_row0(sEx, 0),   ar1 = frag_a_lds_row0(sEx, 32);
    v16h aTZ = frag_a_lds(sTZ, 32, 0, 0);
    for (int nt = 0; nt < 2; ++nt) {
      int n0 = nt * 16;
      v16h B0 = frag_b<PREP>(wsf, OFF_Wqo + nt * 5 + 0, Wqo, 160, n0, 0, 20, 160);
      v16h B1 = frag_b<PREP>(wsf, OFF_Wqo + nt * 5 + 1, Wqo, 160, n0, 32, 20, 160);
      v16h B2 = frag_b<PREP>(wsf, OFF_Wqo + nt * 5 + 2, Wqo, 160, n0, 64, 20, 160);
      v16h B3 = frag_b<PREP>(wsf, OFF_Wqo + nt * 5 + 3, Wqo, 160, n0, 96, 20, 160);
      v16h B4 = frag_b<PREP>(wsf, OFF_Wqo + nt * 5 + 4, Wqo, 160, n0, 128, 20, 160);
      v8f c = frag_c_bias(bqo, n0, 20);
      c = wmma16(ah0, B0, c);
      c = wmma16(ah1, B1, c);
      c = wmma16(ax0, B2, c);
      c = wmma16(ax1, B3, c);
      c = wmma16(aTZ, B4, c);
      v8f c2 = frag_c_bias(bqo, n0, 20);
      c2 = wmma16(ah0, B0, c2);
      c2 = wmma16(ah1, B1, c2);
      c2 = wmma16(ar0, B2, c2);
      c2 = wmma16(ar1, B3, c2);
      c2 = wmma16(aTZ, B4, c2);
      int n = n0 + (lane & 15);
      int mb = (lane >> 4) << 3;
      if (n < 20) {
#pragma unroll
        for (int r = 0; r < 8; ++r) {
          int m = mb + r;
          float v = c[r];
          if (m == 0 || sScen[m] != 0) v = 0.f;           // row0 zero + agent mask
          out[3 * Osz + obase + m * 20 + n] = v;                    // all_qi
          out[0 * Osz + obase + m * 20 + n] = v + sQA[m * 20 + n];  // total
          out[1 * Osz + obase + m * 20 + n] = c2[r];                // aux_q
        }
      }
    }
  }
}

extern "C" void kernel_launch(void* const* d_in, const int* in_sizes, int n_in,
                              void* d_out, int out_size, void* d_ws, size_t ws_size,
                              hipStream_t stream) {
  (void)n_in; (void)out_size;
  const float* inputs   = (const float*)d_in[0];
  const float* onehot   = (const float*)d_in[1];
  const float* task_z   = (const float*)d_in[2];
  const float* hidden   = (const float*)d_in[3];
  const int*   obs_mask = (const int*)d_in[4];
  const int*   scen     = (const int*)d_in[5];
  const float* W1   = (const float*)d_in[6];
  const float* b1   = (const float*)d_in[7];
  const float* Wq   = (const float*)d_in[8];
  const float* bq   = (const float*)d_in[9];
  const float* Wk   = (const float*)d_in[10];
  const float* bk   = (const float*)d_in[11];
  const float* Wv   = (const float*)d_in[12];
  const float* bv   = (const float*)d_in[13];
  const float* Wih  = (const float*)d_in[14];
  const float* Whh  = (const float*)d_in[15];
  const float* bih  = (const float*)d_in[16];
  const float* bhh  = (const float*)d_in[17];
  const float* Wqa  = (const float*)d_in[18];
  const float* bqa  = (const float*)d_in[19];
  const float* Wact = (const float*)d_in[20];
  const float* Wdq  = (const float*)d_in[21];
  const float* bdq  = (const float*)d_in[22];
  const float* Wdk  = (const float*)d_in[23];
  const float* bdk  = (const float*)d_in[24];
  const float* Wdv  = (const float*)d_in[25];
  const float* bdv  = (const float*)d_in[26];
  const float* Wqo  = (const float*)d_in[27];
  const float* bqo  = (const float*)d_in[28];

  const int BS = in_sizes[0] / (NEg * INDIM);
  const bool use_prep = (d_ws != nullptr) && (ws_size >= WS_NEEDED);
  if (use_prep) {
    prep_wfrags<<<dim3(NFRAG), dim3(32), 0, stream>>>(
        W1, Wq, Wk, Wv, Wih, Whh, Wqa, Wact, Wdq, Wdk, Wdv, Wqo, (_Float16*)d_ws);
    scvd_agent_fwd<true><<<dim3(BS), dim3(32), 0, stream>>>(
        inputs, onehot, task_z, hidden, obs_mask, scen,
        W1, b1, Wq, bq, Wk, bk, Wv, bv, Wih, Whh, bih, bhh,
        Wqa, bqa, Wact, Wdq, bdq, Wdk, bdk, Wdv, bdv, Wqo, bqo,
        (const _Float16*)d_ws, (float*)d_out, BS);
  } else {
    scvd_agent_fwd<false><<<dim3(BS), dim3(32), 0, stream>>>(
        inputs, onehot, task_z, hidden, obs_mask, scen,
        W1, b1, Wq, bq, Wk, bk, Wv, bv, Wih, Whh, bih, bhh,
        Wqa, bqa, Wact, Wdq, bdq, Wdk, bdk, Wdv, bdv, Wqo, bqo,
        nullptr, (float*)d_out, BS);
  }
}
